// HybridUltimateBlock_37237366456749
// MI455X (gfx1250) — compile-verified, matrix operand order checked by reading, not run on hardware
//
#include <hip/hip_runtime.h>

// ---------------------------------------------------------------------------
// MI455X (gfx1250, wave32) hybrid block: attention + selective-SSM + gated FFN
// GEMMs: async-copy global->LDS double-buffered, v_wmma_f32_16x16x32_bf16.
// ---------------------------------------------------------------------------

typedef __attribute__((ext_vector_type(16))) __bf16 v16bf;
typedef __attribute__((ext_vector_type(8)))  __bf16 bf16x8;
typedef __attribute__((ext_vector_type(8)))  float  v8f;
typedef __attribute__((ext_vector_type(4)))  float  f32x4;
typedef __attribute__((ext_vector_type(4)))  int    ai4;

#define DEV __device__ __forceinline__

#if defined(__has_builtin)
#  if __has_builtin(__builtin_amdgcn_global_load_async_to_lds_b128) && \
      __has_builtin(__builtin_amdgcn_s_wait_asynccnt)
#    define HAS_ASYNC_CP 1
#  endif
#endif
#ifndef HAS_ASYNC_CP
#  define HAS_ASYNC_CP 0
#endif

DEV __bf16 f2bf(float f) {
  unsigned u = __builtin_bit_cast(unsigned, f);
  unsigned r = u + 0x7FFFu + ((u >> 16) & 1u);           // round-to-nearest-even
  unsigned short h = (unsigned short)(r >> 16);
  return __builtin_bit_cast(__bf16, h);
}

DEV float sigmoidf_(float x) { return 1.0f / (1.0f + __expf(-x)); }

// 16B global -> LDS copy, ASYNCcnt-tracked when the gfx1250 async path exists.
// Builtin signature (from hipcc diagnostic): (int4 AS1*, int4 AS3*, imm, imm).
DEV void cp16_g2l(const __bf16* g, __bf16* l) {
#if HAS_ASYNC_CP
  __builtin_amdgcn_global_load_async_to_lds_b128(
      (__attribute__((address_space(1))) ai4*)g,
      (__attribute__((address_space(3))) ai4*)l, 0, 0);
#else
  *(bf16x8*)l = *(const bf16x8*)g;
#endif
}
DEV void wait_async_lds() {
#if HAS_ASYNC_CP
  __builtin_amdgcn_s_wait_asynccnt(0);
#endif
}

// Build 16-element bf16 WMMA fragment from two contiguous 8-element runs
// (CDNA5 16-bit A/B VGPR layout: K = 8h..8h+7 then 16+8h..16+8h+7).
DEV v16bf make_frag(const __bf16* p0, const __bf16* p1) {
  bf16x8 a = *(const bf16x8*)p0;
  bf16x8 b = *(const bf16x8*)p1;
  v16bf f;
#pragma unroll
  for (int i = 0; i < 8; ++i) { f[i] = a[i]; f[i + 8] = b[i]; }
  return f;
}

// ---------------------------------------------------------------------------
// Weight convert: fp32 [K,N] row-major  ->  bf16 [N,K] (transposed)
// ---------------------------------------------------------------------------
__global__ void wconv_kernel(const float* __restrict__ src, __bf16* __restrict__ dst,
                             int K, int N) {
  size_t idx = (size_t)blockIdx.x * blockDim.x + threadIdx.x;
  size_t tot = (size_t)K * N;
  if (idx >= tot) return;
  int kk = (int)(idx / N), nn = (int)(idx % N);
  dst[(size_t)nn * K + kk] = f2bf(src[idx]);
}

// ---------------------------------------------------------------------------
// Row RMS-norm: out = x * rsqrt(mean(x^2)+eps) * w   (fp32 and/or bf16 out)
// ---------------------------------------------------------------------------
__global__ void rms_kernel(const float* __restrict__ x, const float* __restrict__ w,
                           float* outf, __bf16* outbf, int D) {
  int row = blockIdx.x, tid = threadIdx.x;
  const float* xr = x + (size_t)row * D;
  float ss = 0.f;
  for (int i = tid; i < D; i += blockDim.x) { float v = xr[i]; ss += v * v; }
#pragma unroll
  for (int d = 1; d < 32; d <<= 1) ss += __shfl_xor(ss, d, 32);
  __shared__ float red[8];
  if ((tid & 31) == 0) red[tid >> 5] = ss;
  __syncthreads();
  float tot = 0.f;
#pragma unroll
  for (int i = 0; i < 8; ++i) tot += red[i];
  float sc = rsqrtf(tot / (float)D + 1e-6f);
  for (int i = tid; i < D; i += blockDim.x) {
    float v = xr[i] * sc * w[i];
    if (outf)  outf[(size_t)row * D + i] = v;
    if (outbf) outbf[(size_t)row * D + i] = f2bf(v);
  }
}

// ---------------------------------------------------------------------------
// Tiled bf16 WMMA GEMM with async global->LDS double buffering.
// C[M,N] = A[M,K] @ B[K,N], B given transposed [N,K]. K % 64 == 0.
// Block = 256 thr = 8 waves; tile 64x64; K staged in 64-wide chunks:
//   sA/sB tile = 64x64 bf16 = 8 KB each, two buffers = 32 KB LDS.
// Pipeline per chunk: issue async copies (chunk k+1) -> 4 WMMAs on chunk k
// from LDS -> s_wait_asynccnt 0 -> barrier.
// Epilogue modes: 0 none, 1 sigmoid(z+b), 2 sigmoid((z+b)*g[n]), 3 softplus(z+b)
// ---------------------------------------------------------------------------
__global__ void gemm_bf16(const __bf16* __restrict__ A, const __bf16* __restrict__ BT,
                          float* __restrict__ C, __bf16* __restrict__ Cbf,
                          const float* __restrict__ bias, const float* __restrict__ pregate,
                          int M, int N, int K, int mode) {
  __shared__ __bf16 sA[2][64 * 64];
  __shared__ __bf16 sB[2][64 * 64];
  const int tid = threadIdx.x;
  const int lane = tid & 31, wid = tid >> 5;
  const int lo = lane & 15, hf = lane >> 4;
  const int blockM = blockIdx.y * 64, blockN = blockIdx.x * 64;
  const int wM = (wid >> 1) * 16;   // 0,16,32,48
  const int wN = (wid & 1) * 32;    // 0,32

  // 256 threads x (2 A-chunks + 2 B-chunks) of 16B cover both 8 KB tiles.
  auto issue_copy = [&](int buf, int kk) {
#pragma unroll
    for (int r = 0; r < 2; ++r) {
      int c = tid + r * 256;            // 0..511
      int row = c >> 3, col = (c & 7) * 8;
      cp16_g2l(A + (size_t)(blockM + row) * K + kk + col, &sA[buf][row * 64 + col]);
      cp16_g2l(BT + (size_t)(blockN + row) * K + kk + col, &sB[buf][row * 64 + col]);
    }
  };

  const v8f zero = {0.f, 0.f, 0.f, 0.f, 0.f, 0.f, 0.f, 0.f};
  v8f acc0 = zero, acc1 = zero;

  issue_copy(0, 0);
  wait_async_lds();
  __syncthreads();

  const int nk = K >> 6;
  for (int kc = 0; kc < nk; ++kc) {
    const int cur = kc & 1;
    if (kc + 1 < nk) issue_copy(1 - cur, (kc + 1) << 6);

    const __bf16* la  = &sA[cur][(wM + lo) * 64 + 8 * hf];
    const __bf16* lb0 = &sB[cur][(wN + lo) * 64 + 8 * hf];
    const __bf16* lb1 = &sB[cur][(wN + 16 + lo) * 64 + 8 * hf];
#pragma unroll
    for (int ks = 0; ks < 2; ++ks) {    // two K=32 steps inside the 64-chunk
      v16bf af = make_frag(la + ks * 32, la + ks * 32 + 16);
      v16bf b0 = make_frag(lb0 + ks * 32, lb0 + ks * 32 + 16);
      v16bf b1 = make_frag(lb1 + ks * 32, lb1 + ks * 32 + 16);
      acc0 = __builtin_amdgcn_wmma_f32_16x16x32_bf16(false, af, false, b0, (short)0, acc0, false, false);
      acc1 = __builtin_amdgcn_wmma_f32_16x16x32_bf16(false, af, false, b1, (short)0, acc1, false, false);
    }
    wait_async_lds();
    __syncthreads();
  }

#pragma unroll
  for (int j = 0; j < 8; ++j) {
    int r = blockM + wM + j + 8 * hf;
#pragma unroll
    for (int t = 0; t < 2; ++t) {
      int c = blockN + wN + 16 * t + lo;
      float v = t ? acc1[j] : acc0[j];
      if (bias) v += bias[c];
      if (mode == 1)      v = sigmoidf_(v);
      else if (mode == 2) v = sigmoidf_(v * pregate[c]);
      else if (mode == 3) v = (v > 20.f) ? v : log1pf(__expf(v));
      C[(size_t)r * N + c] = v;
      if (Cbf) Cbf[(size_t)r * N + c] = f2bf(v);
    }
  }
}

// ---------------------------------------------------------------------------
// Per-head RMS(q,k) + RoPE + repack:
//   qbf [B,H,L,hd]   kbf [B,KVH,L,hd]   vT [B,KVH,hd,L]   (all bf16)
// ---------------------------------------------------------------------------
__global__ void qkv_prep(const float* __restrict__ qf, const float* __restrict__ kf,
                         const float* __restrict__ vf,
                         const float* __restrict__ qn_w, const float* __restrict__ kn_w,
                         __bf16* __restrict__ qbf, __bf16* __restrict__ kbf,
                         __bf16* __restrict__ vT) {
  const int row = blockIdx.x, tid = threadIdx.x;
  const int b = row >> 10, pos = row & 1023;
  __shared__ float qs[1024];
  __shared__ float ks[256];

  {  // Q: 16 heads * 64, 4 dims/thread, head == 16 consecutive lanes
    int d0 = tid * 4, hb = d0 & 63;
    f32x4 v = *(const f32x4*)(qf + (size_t)row * 1024 + d0);
    float ss = v[0]*v[0] + v[1]*v[1] + v[2]*v[2] + v[3]*v[3];
#pragma unroll
    for (int d = 1; d < 16; d <<= 1) ss += __shfl_xor(ss, d, 32);
    float sc = rsqrtf(ss / 64.f + 1e-6f);
#pragma unroll
    for (int c = 0; c < 4; ++c) qs[d0 + c] = v[c] * sc * qn_w[hb + c];
  }
  if (tid < 64) {  // K + V
    int d0 = tid * 4, head = tid >> 4, hb = d0 & 63;
    f32x4 v = *(const f32x4*)(kf + (size_t)row * 256 + d0);
    float ss = v[0]*v[0] + v[1]*v[1] + v[2]*v[2] + v[3]*v[3];
#pragma unroll
    for (int d = 1; d < 16; d <<= 1) ss += __shfl_xor(ss, d, 32);
    float sc = rsqrtf(ss / 64.f + 1e-6f);
#pragma unroll
    for (int c = 0; c < 4; ++c) ks[d0 + c] = v[c] * sc * kn_w[hb + c];
    f32x4 vv = *(const f32x4*)(vf + (size_t)row * 256 + d0);
#pragma unroll
    for (int c = 0; c < 4; ++c)
      vT[(((size_t)(b * 4 + head)) * 64 + (hb + c)) * 1024 + pos] = f2bf(vv[c]);
  }
  __syncthreads();

  const float NEG = -0.28782313662425572f;  // -2*ln(10000)/64
  {  // RoPE on q
    int d0 = tid * 4, head = tid >> 4, hb = d0 & 63;
#pragma unroll
    for (int c = 0; c < 4; ++c) {
      int hd_i = hb + c;
      int i2 = (hd_i < 32) ? hd_i : hd_i - 32;
      float ang = (float)pos * __expf((float)i2 * NEG);
      float sn, cs; __sincosf(ang, &sn, &cs);
      float a = qs[head * 64 + hd_i];
      float r = (hd_i < 32) ? -qs[head * 64 + hd_i + 32] : qs[head * 64 + hd_i - 32];
      qbf[(((size_t)(b * 16 + head)) * 1024 + pos) * 64 + hd_i] = f2bf(a * cs + r * sn);
    }
  }
  if (tid < 64) {  // RoPE on k
    int d0 = tid * 4, head = tid >> 4, hb = d0 & 63;
#pragma unroll
    for (int c = 0; c < 4; ++c) {
      int hd_i = hb + c;
      int i2 = (hd_i < 32) ? hd_i : hd_i - 32;
      float ang = (float)pos * __expf((float)i2 * NEG);
      float sn, cs; __sincosf(ang, &sn, &cs);
      float a = ks[head * 64 + hd_i];
      float r = (hd_i < 32) ? -ks[head * 64 + hd_i + 32] : ks[head * 64 + hd_i - 32];
      kbf[(((size_t)(b * 4 + head)) * 1024 + pos) * 64 + hd_i] = f2bf(a * cs + r * sn);
    }
  }
}

// ---------------------------------------------------------------------------
// Flash-style GQA attention (non-causal, L=1024, hd=64). One wave per 16 query
// rows. QK^T and P@V via WMMA bf16; online softmax with half-wave shuffles;
// P re-fragmented through LDS. out = [B,L,H,hd] bf16.
// ---------------------------------------------------------------------------
__global__ void attn_kernel(const __bf16* __restrict__ q, const __bf16* __restrict__ k,
                            const __bf16* __restrict__ vT, __bf16* __restrict__ out) {
  const int bh = blockIdx.y, b = bh >> 4, h = bh & 15, kvh = h >> 2;
  const int l0 = blockIdx.x * 16;
  const int lane = threadIdx.x, lo = lane & 15, hf = lane >> 4;
  __shared__ float pld[16 * 32];

  const __bf16* qr = q + (((size_t)(b * 16 + h)) * 1024 + (l0 + lo)) * 64;
  v16bf aq0 = make_frag(qr + 8 * hf,       qr + 16 + 8 * hf);   // hd 0..31
  v16bf aq1 = make_frag(qr + 32 + 8 * hf,  qr + 48 + 8 * hf);   // hd 32..63
  const __bf16* kbase = k  + ((size_t)(b * 4 + kvh)) * 1024 * 64;
  const __bf16* vbase = vT + ((size_t)(b * 4 + kvh)) * 64 * 1024;

  const v8f zero = {0.f, 0.f, 0.f, 0.f, 0.f, 0.f, 0.f, 0.f};
  v8f acc[4] = {zero, zero, zero, zero};
  float rowm[8], rowl[8];
#pragma unroll
  for (int j = 0; j < 8; ++j) { rowm[j] = -3.0e38f; rowl[j] = 0.f; }

  for (int kt = 0; kt < 1024; kt += 32) {
    __builtin_prefetch(kbase + (size_t)(kt + 64) * 64, 0, 1);
    const __bf16* kr0 = kbase + (size_t)(kt + lo) * 64;
    const __bf16* kr1 = kbase + (size_t)(kt + 16 + lo) * 64;
    v16bf bk00 = make_frag(kr0 + 8 * hf,      kr0 + 16 + 8 * hf);
    v16bf bk01 = make_frag(kr0 + 32 + 8 * hf, kr0 + 48 + 8 * hf);
    v16bf bk10 = make_frag(kr1 + 8 * hf,      kr1 + 16 + 8 * hf);
    v16bf bk11 = make_frag(kr1 + 32 + 8 * hf, kr1 + 48 + 8 * hf);
    v8f s0 = zero, s1 = zero;
    s0 = __builtin_amdgcn_wmma_f32_16x16x32_bf16(false, aq0, false, bk00, (short)0, s0, false, false);
    s0 = __builtin_amdgcn_wmma_f32_16x16x32_bf16(false, aq1, false, bk01, (short)0, s0, false, false);
    s1 = __builtin_amdgcn_wmma_f32_16x16x32_bf16(false, aq0, false, bk10, (short)0, s1, false, false);
    s1 = __builtin_amdgcn_wmma_f32_16x16x32_bf16(false, aq1, false, bk11, (short)0, s1, false, false);

#pragma unroll
    for (int j = 0; j < 8; ++j) {
      float a0 = s0[j] * 0.125f, a1 = s1[j] * 0.125f;   // 1/sqrt(64)
      float mx = fmaxf(a0, a1);
#pragma unroll
      for (int d = 1; d < 16; d <<= 1) mx = fmaxf(mx, __shfl_xor(mx, d, 32));
      float nm = fmaxf(rowm[j], mx);
      float p0 = __expf(a0 - nm), p1 = __expf(a1 - nm);
      float sm = p0 + p1;
#pragma unroll
      for (int d = 1; d < 16; d <<= 1) sm += __shfl_xor(sm, d, 32);
      float rs = __expf(rowm[j] - nm);
      rowl[j] = rowl[j] * rs + sm;
      rowm[j] = nm;
#pragma unroll
      for (int t = 0; t < 4; ++t) acc[t][j] *= rs;
      pld[(j + 8 * hf) * 32 + lo] = p0;
      pld[(j + 8 * hf) * 32 + 16 + lo] = p1;
    }
    __syncthreads();
    v16bf ap;
#pragma unroll
    for (int i = 0; i < 8; ++i) {
      ap[i]     = f2bf(pld[lo * 32 + 8 * hf + i]);
      ap[i + 8] = f2bf(pld[lo * 32 + 16 + 8 * hf + i]);
    }
    __syncthreads();
#pragma unroll
    for (int t = 0; t < 4; ++t) {
      const __bf16* vr = vbase + (size_t)(t * 16 + lo) * 1024 + kt;
      v16bf bv = make_frag(vr + 8 * hf, vr + 16 + 8 * hf);
      acc[t] = __builtin_amdgcn_wmma_f32_16x16x32_bf16(false, ap, false, bv, (short)0, acc[t], false, false);
    }
  }

#pragma unroll
  for (int j = 0; j < 8; ++j) {
    int r = l0 + j + 8 * hf;
    float inv = 1.0f / rowl[j];
#pragma unroll
    for (int t = 0; t < 4; ++t) {
      int c = t * 16 + lo;
      out[(((size_t)b * 1024 + r) * 16 + h) * 64 + c] = f2bf(acc[t][j] * inv);
    }
  }
}

// ---------------------------------------------------------------------------
// Selective scan: one wave per (b,d) row; 128-wide state, 4 elems/lane.
// ---------------------------------------------------------------------------
__global__ void scan_kernel(const float* __restrict__ delta, const float* __restrict__ Bs,
                            const float* __restrict__ Cs, const float* __restrict__ xn,
                            const float* __restrict__ A_log, float* __restrict__ y) {
  const int wid = threadIdx.x >> 5, lane = threadIdx.x & 31;
  const int idx = blockIdx.x * 8 + wid;          // 0..2047 = (b,d)
  const int b = idx >> 10, d = idx & 1023;
  float Aq[4], hq[4];
#pragma unroll
  for (int i = 0; i < 4; ++i) {
    Aq[i] = -__expf(A_log[(size_t)d * 128 + lane * 4 + i]);
    hq[i] = 0.f;
  }
  for (int t = 0; t < 1024; ++t) {
    size_t r = (size_t)b * 1024 + t;
    float dt = delta[r * 1024 + d];
    float xt = xn[r * 1024 + d];
    f32x4 Bv = *(const f32x4*)(Bs + r * 128 + lane * 4);
    f32x4 Cv = *(const f32x4*)(Cs + r * 128 + lane * 4);
    float part = 0.f;
#pragma unroll
    for (int i = 0; i < 4; ++i) {
      float dA = __expf(dt * Aq[i]);
      hq[i] = dA * hq[i] + dt * Bv[i] * xt;
      part += hq[i] * Cv[i];
    }
#pragma unroll
    for (int d2 = 1; d2 < 32; d2 <<= 1) part += __shfl_xor(part, d2, 32);
    if (lane == 0) y[r * 1024 + d] = part;
  }
}

// ---------------------------------------------------------------------------
// Elementwise glue kernels
// ---------------------------------------------------------------------------
__global__ void ew_mul(const float* a, const float* b, float* outf, __bf16* outbf, size_t n) {
  for (size_t i = (size_t)blockIdx.x * blockDim.x + threadIdx.x; i < n;
       i += (size_t)gridDim.x * blockDim.x) {
    float v = a[i] * b[i];
    if (outf)  outf[i] = v;
    if (outbf) outbf[i] = f2bf(v);
  }
}
__global__ void ew_drbc(const float* __restrict__ xp, const float* __restrict__ imp,
                        __bf16* __restrict__ drbf, float* __restrict__ Bso,
                        float* __restrict__ Cso, size_t M) {
  for (size_t i = (size_t)blockIdx.x * blockDim.x + threadIdx.x; i < M * 128;
       i += (size_t)gridDim.x * blockDim.x) {
    size_t r = i >> 7; int n = (int)(i & 127);
    float iv = imp[i];
    Bso[i] = xp[r * 320 + 64 + n] * iv;
    Cso[i] = xp[r * 320 + 192 + n] * iv;
    if (n < 64) drbf[r * 64 + n] = f2bf(xp[r * 320 + n]);
  }
}
__global__ void ew_ssmpre(const float* __restrict__ y, const float* __restrict__ x,
                          const float* __restrict__ Dp, __bf16* __restrict__ out, size_t n) {
  for (size_t i = (size_t)blockIdx.x * blockDim.x + threadIdx.x; i < n;
       i += (size_t)gridDim.x * blockDim.x)
    out[i] = f2bf(y[i] + x[i] * Dp[i & 1023]);
}
__global__ void ew_combine(const float* __restrict__ x, const float* __restrict__ at,
                           const float* __restrict__ ss, float* __restrict__ x1, size_t n) {
  for (size_t i = (size_t)blockIdx.x * blockDim.x + threadIdx.x; i < n;
       i += (size_t)gridDim.x * blockDim.x)
    x1[i] = x[i] + 0.5f * at[i] + 0.5f * ss[i];
}
__global__ void ew_ffnact(const float* __restrict__ u, __bf16* __restrict__ out, size_t M) {
  for (size_t i = (size_t)blockIdx.x * blockDim.x + threadIdx.x; i < M * 4096;
       i += (size_t)gridDim.x * blockDim.x) {
    size_t r = i >> 12; int c = (int)(i & 4095);
    float a = u[r * 8192 + c], g = u[r * 8192 + 4096 + c];
    out[i] = f2bf((a * sigmoidf_(a)) * g);
  }
}
__global__ void ew_add(const float* a, const float* b, float* out, size_t n) {
  for (size_t i = (size_t)blockIdx.x * blockDim.x + threadIdx.x; i < n;
       i += (size_t)gridDim.x * blockDim.x)
    out[i] = a[i] + b[i];
}

// ---------------------------------------------------------------------------
// Host-side orchestration
// ---------------------------------------------------------------------------
extern "C" void kernel_launch(void* const* d_in, const int* in_sizes, int n_in,
                              void* d_out, int out_size, void* d_ws, size_t ws_size,
                              hipStream_t stream) {
  (void)in_sizes; (void)n_in; (void)out_size; (void)ws_size;
  const int M = 2048, D = 1024;  // M = B*L
  const size_t MD = (size_t)M * D;

  const float* x      = (const float*)d_in[0];
  const float* naw    = (const float*)d_in[1];
  const float* Wq     = (const float*)d_in[2];
  const float* Wk     = (const float*)d_in[3];
  const float* Wv     = (const float*)d_in[4];
  const float* Wo     = (const float*)d_in[5];
  const float* qn_w   = (const float*)d_in[6];
  const float* kn_w   = (const float*)d_in[7];
  const float* snw    = (const float*)d_in[8];
  const float* Wx     = (const float*)d_in[9];
  const float* Wdt    = (const float*)d_in[10];
  const float* bdt    = (const float*)d_in[11];
  const float* A_log  = (const float*)d_in[12];
  const float* Dp     = (const float*)d_in[13];
  const float* Wsel   = (const float*)d_in[14];
  const float* bsel   = (const float*)d_in[15];
  const float* selg   = (const float*)d_in[16];
  const float* Wimp   = (const float*)d_in[17];
  const float* bimp   = (const float*)d_in[18];
  const float* Wg     = (const float*)d_in[19];
  const float* bg     = (const float*)d_in[20];
  const float* Wout   = (const float*)d_in[21];
  const float* fnw    = (const float*)d_in[22];
  const float* W1     = (const float*)d_in[23];
  const float* W2     = (const float*)d_in[24];
  float* out = (float*)d_out;

  char* ws = (char*)d_ws;
  size_t off = 0;
  auto alloc = [&](size_t bytes) -> char* {
    char* p = ws + off; off += (bytes + 255) & ~(size_t)255; return p;
  };

  // bf16-transposed weights
  __bf16* WqT  = (__bf16*)alloc((size_t)D * D * 2);
  __bf16* WkT  = (__bf16*)alloc((size_t)D * 256 * 2);
  __bf16* WvT  = (__bf16*)alloc((size_t)D * 256 * 2);
  __bf16* WoT  = (__bf16*)alloc((size_t)D * D * 2);
  __bf16* WxT  = (__bf16*)alloc((size_t)D * 320 * 2);
  __bf16* WdtT = (__bf16*)alloc((size_t)64 * 1024 * 2);
  __bf16* WselT= (__bf16*)alloc((size_t)D * D * 2);
  __bf16* WimpT= (__bf16*)alloc((size_t)D * 128 * 2);
  __bf16* WgT  = (__bf16*)alloc((size_t)D * D * 2);
  __bf16* WoutT= (__bf16*)alloc((size_t)D * D * 2);
  __bf16* W1T  = (__bf16*)alloc((size_t)D * 8192 * 2);
  __bf16* W2T  = (__bf16*)alloc((size_t)4096 * D * 2);

  // activations
  __bf16* xa_bf  = (__bf16*)alloc(MD * 2);
  float*  qf     = (float*) alloc(MD * 4);
  float*  kf     = (float*) alloc((size_t)M * 256 * 4);
  float*  vf     = (float*) alloc((size_t)M * 256 * 4);
  __bf16* qbf    = (__bf16*)alloc(MD * 2);
  __bf16* kbf    = (__bf16*)alloc((size_t)M * 256 * 2);
  __bf16* vTbf   = (__bf16*)alloc((size_t)M * 256 * 2);
  __bf16* araw   = (__bf16*)alloc(MD * 2);
  float*  attn_o = (float*) alloc(MD * 4);
  float*  xn0f   = (float*) alloc(MD * 4);
  __bf16* xn0bf  = (__bf16*)alloc(MD * 2);
  float*  xp     = (float*) alloc((size_t)M * 320 * 4);
  __bf16* dr_bf  = (__bf16*)alloc((size_t)M * 64 * 2);
  float*  sel    = (float*) alloc(MD * 4);
  float*  xn1f   = (float*) alloc(MD * 4);
  __bf16* xn1bf  = (__bf16*)alloc(MD * 2);
  float*  imp    = (float*) alloc((size_t)M * 128 * 4);
  float*  delta  = (float*) alloc(MD * 4);
  float*  gate   = (float*) alloc(MD * 4);
  float*  xn2    = (float*) alloc(MD * 4);
  float*  Bsb    = (float*) alloc((size_t)M * 128 * 4);
  float*  Csb    = (float*) alloc((size_t)M * 128 * 4);
  float*  yb     = (float*) alloc(MD * 4);
  __bf16* ssmin  = (__bf16*)alloc(MD * 2);
  float*  ssm_o  = (float*) alloc(MD * 4);
  float*  x1     = (float*) alloc(MD * 4);
  __bf16* xf_bf  = (__bf16*)alloc(MD * 2);
  float*  u      = (float*) alloc((size_t)M * 8192 * 4);
  __bf16* ffin   = (__bf16*)alloc((size_t)M * 4096 * 2);
  float*  ff     = (float*) alloc(MD * 4);

  auto wconv = [&](const float* src, __bf16* dst, int K, int N) {
    size_t tot = (size_t)K * N;
    wconv_kernel<<<(unsigned)((tot + 255) / 256), 256, 0, stream>>>(src, dst, K, N);
  };
  auto gemm = [&](const __bf16* A, const __bf16* BT, float* C, __bf16* Cbf,
                  const float* bias, const float* pg, int Mm, int Nn, int Kk, int mode) {
    dim3 g(Nn / 64, Mm / 64);
    gemm_bf16<<<g, 256, 0, stream>>>(A, BT, C, Cbf, bias, pg, Mm, Nn, Kk, mode);
  };

  // Weight conversion (K x N -> bf16 N x K)
  wconv(Wq, WqT, 1024, 1024);   wconv(Wk, WkT, 1024, 256);
  wconv(Wv, WvT, 1024, 256);    wconv(Wo, WoT, 1024, 1024);
  wconv(Wx, WxT, 1024, 320);    wconv(Wdt, WdtT, 64, 1024);
  wconv(Wsel, WselT, 1024, 1024); wconv(Wimp, WimpT, 1024, 128);
  wconv(Wg, WgT, 1024, 1024);   wconv(Wout, WoutT, 1024, 1024);
  wconv(W1, W1T, 1024, 8192);   wconv(W2, W2T, 4096, 1024);

  // ----- attention branch -----
  rms_kernel<<<M, 256, 0, stream>>>(x, naw, nullptr, xa_bf, D);
  gemm(xa_bf, WqT, qf, nullptr, nullptr, nullptr, M, 1024, 1024, 0);
  gemm(xa_bf, WkT, kf, nullptr, nullptr, nullptr, M, 256, 1024, 0);
  gemm(xa_bf, WvT, vf, nullptr, nullptr, nullptr, M, 256, 1024, 0);
  qkv_prep<<<M, 256, 0, stream>>>(qf, kf, vf, qn_w, kn_w, qbf, kbf, vTbf);
  attn_kernel<<<dim3(64, 32), 32, 0, stream>>>(qbf, kbf, vTbf, araw);
  gemm(araw, WoT, attn_o, nullptr, nullptr, nullptr, M, 1024, 1024, 0);

  // ----- SSM branch -----
  rms_kernel<<<M, 256, 0, stream>>>(x, snw, xn0f, xn0bf, D);
  gemm(xn0bf, WxT, xp, nullptr, nullptr, nullptr, M, 320, 1024, 0);
  gemm(xn0bf, WselT, sel, nullptr, bsel, selg, M, 1024, 1024, 2);
  ew_mul<<<2048, 256, 0, stream>>>(xn0f, sel, xn1f, xn1bf, MD);
  gemm(xn1bf, WimpT, imp, nullptr, bimp, nullptr, M, 128, 1024, 1);
  ew_drbc<<<2048, 256, 0, stream>>>(xp, imp, dr_bf, Bsb, Csb, (size_t)M);
  gemm(dr_bf, WdtT, delta, nullptr, bdt, nullptr, M, 1024, 64, 3);
  gemm(xn1bf, WgT, gate, nullptr, bg, nullptr, M, 1024, 1024, 1);
  ew_mul<<<2048, 256, 0, stream>>>(xn1f, gate, xn2, nullptr, MD);
  scan_kernel<<<256, 256, 0, stream>>>(delta, Bsb, Csb, xn2, A_log, yb);
  ew_ssmpre<<<2048, 256, 0, stream>>>(yb, x, Dp, ssmin, MD);
  gemm(ssmin, WoutT, ssm_o, nullptr, nullptr, nullptr, M, 1024, 1024, 0);

  // ----- combine + FFN -----
  ew_combine<<<2048, 256, 0, stream>>>(x, attn_o, ssm_o, x1, MD);
  rms_kernel<<<M, 256, 0, stream>>>(x1, fnw, nullptr, xf_bf, D);
  gemm(xf_bf, W1T, u, nullptr, nullptr, nullptr, M, 8192, 1024, 0);
  ew_ffnact<<<4096, 256, 0, stream>>>(u, ffin, (size_t)M);
  gemm(ffin, W2T, ff, nullptr, nullptr, nullptr, M, 1024, 4096, 0);
  ew_add<<<2048, 256, 0, stream>>>(x1, ff, out, MD);
}
